// MultiHeadedAttentionBlock_60258391163160
// MI455X (gfx1250) — matrix operand order repaired and checked by reading, not compiled, not run on hardware
//
#include <hip/hip_runtime.h>

// ---------------------------------------------------------------------------
// MI455X (gfx1250) MHA block: bf16 WMMA (v_wmma_f32_16x16x32_bf16) everywhere,
// TDM (tensor_load_to_lds) async staging for bf16 tiles, wave32, WGP-mode.
// ---------------------------------------------------------------------------

typedef __attribute__((ext_vector_type(16))) __bf16 v16bf;
typedef __attribute__((ext_vector_type(8)))  float  v8f;
typedef __attribute__((ext_vector_type(4)))  unsigned int v4u;
typedef __attribute__((ext_vector_type(8)))  int    v8i_;
typedef __attribute__((ext_vector_type(4)))  int    v4i_;

#define B_  4
#define S_  1024
#define D_  1024
#define H_  16
#define HD_ 64

__device__ __forceinline__ __bf16 cvt_bf16(float f) { return (__bf16)f; }

// ---------------------------------------------------------------------------
// TDM: 2D tile DMA global -> LDS. D# per CDNA5 ISA §8.3/8.4.
// tensor dims == tile dims (no OOB), stride0 = row pitch (elems), bf16 data.
// Must be executed by a single wave; wave-uniform args. Not in a cluster, so
// workgroup_mask = 0. Tracked on TENSORcnt.
// ---------------------------------------------------------------------------
__device__ __forceinline__ void tdm_load_2d_bf16(
    unsigned lds_off, const void* gptr,
    unsigned tile_w, unsigned tile_h, unsigned stride_elems)
{
    unsigned long long ga = (unsigned long long)(size_t)gptr;
    v4u g0;
    g0[0] = 1u;                                    // count=1, user descriptor
    g0[1] = lds_off;                               // LDS byte address
    g0[2] = (unsigned)(ga & 0xFFFFFFFFu);          // global_addr[31:0]
    g0[3] = (unsigned)((ga >> 32) & 0x1FFFFFFu)    // global_addr[56:32]
          | (2u << 30);                            // type = 2 ("image")
    unsigned long long st0 = (unsigned long long)stride_elems;
    v8i_ g1;
    g1[0] = (int)(1u << 16);                       // wg_mask=0, data_size=1 (2B)
    g1[1] = (int)((tile_w & 0xFFFFu) << 16);       // tensor_dim0[15:0]
    g1[2] = (int)(((tile_w >> 16) & 0xFFFFu)       // tensor_dim0[31:16]
          | ((tile_h & 0xFFFFu) << 16));           // tensor_dim1[15:0]
    g1[3] = (int)(((tile_h >> 16) & 0xFFFFu)       // tensor_dim1[31:16]
          | ((tile_w & 0xFFFFu) << 16));           // tile_dim0
    g1[4] = (int)(tile_h & 0xFFFFu);               // tile_dim1; tile_dim2=0
    g1[5] = (int)(unsigned)(st0 & 0xFFFFFFFFull);  // tensor_dim0_stride[31:0]
    g1[6] = (int)(unsigned)((st0 >> 32) & 0xFFFFull); // stride0[47:32]
    g1[7] = 0;                                     // tensor_dim1_stride hi (2D)
#if __clang_major__ >= 23
    __builtin_amdgcn_tensor_load_to_lds(g0, g1, (v4i_)0, (v4i_)0, (v8i_)0, 0);
#else
    __builtin_amdgcn_tensor_load_to_lds(g0, g1, (v4i_)0, (v4i_)0, 0);
#endif
}

// --- A-fragment (16 x K=32), source row-major [row][k], stride in elems.
__device__ __forceinline__ v16bf load_a_frag(const __bf16* p0, int stride) {
    const int lane = threadIdx.x & 31, lr = lane & 15, grp = lane >> 4;
    const __bf16* p = p0 + (size_t)lr * stride + grp * 16;
    v16bf a;
#pragma unroll
    for (int i = 0; i < 8; ++i) a[i]     = p[i];
#pragma unroll
    for (int i = 0; i < 8; ++i) a[8 + i] = p[8 + i];
    return a;
}

// --- B-fragment (K=32 x 16), source layout [n][k] (k contiguous).
__device__ __forceinline__ v16bf load_b_frag_rowk(const __bf16* p0, int stride) {
    const int lane = threadIdx.x & 31, lr = lane & 15, grp = lane >> 4;
    const __bf16* p = p0 + (size_t)lr * stride + grp * 16;
    v16bf b;
#pragma unroll
    for (int i = 0; i < 16; ++i) b[i] = p[i];
    return b;
}

// --- B-fragment, source layout [k][n] (n contiguous, k strided).
__device__ __forceinline__ v16bf load_b_frag_colk(const __bf16* p0, int kstride) {
    const int lane = threadIdx.x & 31, lr = lane & 15, grp = lane >> 4;
    const __bf16* p = p0 + (size_t)(grp * 16) * kstride + lr;
    v16bf b;
#pragma unroll
    for (int i = 0; i < 16; ++i) b[i] = p[(size_t)i * kstride];
    return b;
}

__device__ __forceinline__ v8f wmma_bf16(v16bf a, v16bf b, v8f c) {
    return __builtin_amdgcn_wmma_f32_16x16x32_bf16(false, a, false, b,
                                                   (short)0, c, false, false);
}

// ---------------------------------------------------------------------------
// Projection: out = x @ w.T (row-masked), f32 in -> bf16 out in head layout.
// Block 256 = 8 waves. BM=64, BN=128, BK=32. Wave tile 2x2 (4 WMMA/step).
// (f32 source needs convert, so staging stays on the VALU cvt path.)
// ---------------------------------------------------------------------------
__global__ __launch_bounds__(256) void proj_bf16_kernel(
    const float* __restrict__ x, const float* __restrict__ w,
    const unsigned char* __restrict__ mask, __bf16* __restrict__ out_heads)
{
    __shared__ __bf16 As[64 * 32];    // [m][k]
    __shared__ __bf16 Bs[128 * 32];   // [n][k]  (== w.T as [k][n])
    const int m0 = blockIdx.x * 64, n0 = blockIdx.y * 128;
    const int t = threadIdx.x;
    const int wave = t >> 5, lane = t & 31, lr = lane & 15, grp = lane >> 4;
    const int mt = (wave >> 2) * 2;
    const int nt = (wave & 3) * 2;
    const int arow = t >> 2, acol = (t & 3) * 8;
    const int brow = t >> 1, bcol = (t & 1) * 16;

    v8f acc00 = {}, acc01 = {}, acc10 = {}, acc11 = {};
    for (int kk = 0; kk < D_; kk += 32) {
        const float* xp = x + (size_t)(m0 + arow) * D_ + kk + acol;
        const float* wp = w + (size_t)(n0 + brow) * D_ + kk + bcol;
        __builtin_prefetch(xp + 32, 0, 1);   // -> global_prefetch_b8
        __builtin_prefetch(wp + 32, 0, 1);
#pragma unroll
        for (int j = 0; j < 8; ++j)  As[arow * 32 + acol + j] = cvt_bf16(xp[j]);
#pragma unroll
        for (int j = 0; j < 16; ++j) Bs[brow * 32 + bcol + j] = cvt_bf16(wp[j]);
        __syncthreads();
        v16bf b0 = load_b_frag_rowk(Bs + (nt + 0) * 16 * 32, 32);
        v16bf b1 = load_b_frag_rowk(Bs + (nt + 1) * 16 * 32, 32);
        v16bf a0 = load_a_frag(As + (mt + 0) * 16 * 32, 32);
        v16bf a1 = load_a_frag(As + (mt + 1) * 16 * 32, 32);
        acc00 = wmma_bf16(a0, b0, acc00);
        acc01 = wmma_bf16(a0, b1, acc01);
        acc10 = wmma_bf16(a1, b0, acc10);
        acc11 = wmma_bf16(a1, b1, acc11);
        __syncthreads();
    }
    auto store_tile = [&](const v8f& a, int tm, int tn) {
        const int n = n0 + tn * 16 + lr;
        const int h = n >> 6, hd = n & 63;
#pragma unroll
        for (int r = 0; r < 8; ++r) {
            int gm = m0 + tm * 16 + r + grp * 8;
            float v = mask[gm] ? a[r] : 0.0f;
            int bb = gm >> 10, ss = gm & 1023;
            out_heads[(((size_t)(bb * H_ + h)) * S_ + ss) * HD_ + hd] = cvt_bf16(v);
        }
    };
    store_tile(acc00, mt + 0, nt + 0);
    store_tile(acc01, mt + 0, nt + 1);
    store_tile(acc10, mt + 1, nt + 0);
    store_tile(acc11, mt + 1, nt + 1);
}

// ---------------------------------------------------------------------------
// Attention: one WG per (b, h, 16-query-row tile). 256 threads = 8 waves.
// TDM pre-stages the whole per-head V (1024x64 bf16, 128KB) into LDS at
// kernel entry; the DMA overlaps scores+softmax and is drained with
// s_wait_tensorcnt before the P@V phase.
// LDS: 64KB scores + 32KB P + 4KB partials + 128KB V = 228KB (<320KB/WGP).
// ---------------------------------------------------------------------------
__global__ __launch_bounds__(256) void attn_kernel(
    const __bf16* __restrict__ Qh, const __bf16* __restrict__ Kh,
    const __bf16* __restrict__ Vh, const float* __restrict__ ew,
    const unsigned char* __restrict__ qm, const unsigned char* __restrict__ km,
    __bf16* __restrict__ ctx)
{
    __shared__ float  sc[16 * S_];
    __shared__ __bf16 pb[16 * S_];
    __shared__ float  cpart[4 * 16 * 16];
    __shared__ __bf16 Vs[S_ * HD_];          // TDM-staged V for this head

    const int b = blockIdx.z, h = blockIdx.y, qt = blockIdx.x;
    const int s0 = qt * 16;
    const int t = threadIdx.x, wave = t >> 5, lane = t & 31;
    const int lr = lane & 15, grp = lane >> 4;
    const size_t headoff = ((size_t)(b * H_ + h)) * S_ * HD_;
    const float NEG = -__builtin_inff();

    // ---- kick off async V DMA (wave 0 only; overlaps phases 1-2)
    if (wave == 0) {
        tdm_load_2d_bf16((unsigned)(size_t)&Vs[0], Vh + headoff,
                         /*tile_w=*/HD_, /*tile_h=*/S_, /*stride=*/HD_);
    }

    // Q A-fragments (hd 0..31, 32..63) held in VGPRs for all key tiles
    const __bf16* Qp = Qh + headoff + (size_t)s0 * HD_;
    v16bf qa0 = load_a_frag(Qp, HD_);
    v16bf qa1 = load_a_frag(Qp + 32, HD_);

    // ---- phase 1: scores; wave owns keys [wave*128, wave*128+128)
    for (int j = 0; j < 8; ++j) {
        const int key0 = wave * 128 + j * 16;
        v16bf kb0 = load_b_frag_rowk(Kh + headoff + (size_t)key0 * HD_, HD_);
        v16bf kb1 = load_b_frag_rowk(Kh + headoff + (size_t)key0 * HD_ + 32, HD_);
        v8f a = {};
        a = wmma_bf16(qa0, kb0, a);
        a = wmma_bf16(qa1, kb1, a);
        const int key = key0 + lr;
        const bool kvm = km[b * S_ + key] != 0;
#pragma unroll
        for (int r = 0; r < 8; ++r) {
            const int qrow = r + grp * 8;
            const int gq = s0 + qrow;
            float v = a[r] * 0.03125f * ew[((size_t)b * S_ + gq) * S_ + key];
            bool m = kvm && (qm[b * S_ + gq] != 0);
            sc[qrow * S_ + key] = m ? v : NEG;
        }
    }
    __syncthreads();

    // ---- phase 2: softmax over each of 16 rows; wave handles 2 rows
    for (int rr = 0; rr < 2; ++rr) {
        const int row = wave * 2 + rr;
        float mx = NEG;
        for (int k2 = lane; k2 < S_; k2 += 32) mx = fmaxf(mx, sc[row * S_ + k2]);
#pragma unroll
        for (int off = 16; off > 0; off >>= 1) mx = fmaxf(mx, __shfl_xor(mx, off, 32));
        float sum = 0.0f;
        for (int k2 = lane; k2 < S_; k2 += 32) {
            float xv = sc[row * S_ + k2];
            float e = (mx == NEG) ? 0.0f : __expf(xv - mx);
            sc[row * S_ + k2] = e;
            sum += e;
        }
#pragma unroll
        for (int off = 16; off > 0; off >>= 1) sum += __shfl_xor(sum, off, 32);
        const float inv = (sum > 0.0f) ? (1.0f / sum) : 0.0f;
        for (int k2 = lane; k2 < S_; k2 += 32)
            pb[row * S_ + k2] = cvt_bf16(sc[row * S_ + k2] * inv);
    }
    // drain V DMA before anyone reads Vs, then barrier releases all waves
    if (wave == 0) __builtin_amdgcn_s_wait_tensorcnt(0);
    __syncthreads();

    // ---- phase 3: ctx = P @ V (V from LDS). ct = hd tile, kh = key half.
    const int ct = wave & 3;
    const int kh = wave >> 2;
    v8f acc = {};
    for (int j = 0; j < 16; ++j) {
        const int key0 = kh * 512 + j * 32;
        v16bf pa = load_a_frag(pb + key0, S_);
        v16bf vb = load_b_frag_colk(Vs + (size_t)key0 * HD_ + ct * 16, HD_);
        acc = wmma_bf16(pa, vb, acc);
    }
    if (kh == 1) {
#pragma unroll
        for (int r = 0; r < 8; ++r)
            cpart[ct * 256 + (r + grp * 8) * 16 + lr] = acc[r];
    }
    __syncthreads();
    if (kh == 0) {
#pragma unroll
        for (int r = 0; r < 8; ++r) {
            const int m = r + grp * 8;
            float v = acc[r] + cpart[ct * 256 + m * 16 + lr];
            ctx[((size_t)(b * S_ + s0 + m)) * D_ + h * HD_ + ct * 16 + lr] = cvt_bf16(v);
        }
    }
}

// ---------------------------------------------------------------------------
// Output projection: out = ctx(bf16) @ wo.T + bo, row-masked, f32 out.
// A-tile (bf16) staged per K-step by TDM; B-tile converted f32->bf16 by VALU.
// ---------------------------------------------------------------------------
__global__ __launch_bounds__(256) void out_proj_kernel(
    const __bf16* __restrict__ ctxb, const float* __restrict__ w,
    const float* __restrict__ bias, const unsigned char* __restrict__ mask,
    float* __restrict__ out)
{
    __shared__ __bf16 As[64 * 32];
    __shared__ __bf16 Bs[128 * 32];
    const int m0 = blockIdx.x * 64, n0 = blockIdx.y * 128;
    const int t = threadIdx.x;
    const int wave = t >> 5, lane = t & 31, lr = lane & 15, grp = lane >> 4;
    const int mt = (wave >> 2) * 2;
    const int nt = (wave & 3) * 2;
    const int brow = t >> 1, bcol = (t & 1) * 16;

    v8f acc00 = {}, acc01 = {}, acc10 = {}, acc11 = {};
    for (int kk = 0; kk < D_; kk += 32) {
        // A tile 64x32 bf16 via TDM (wave 0), B tile converted by all waves
        if (wave == 0) {
            tdm_load_2d_bf16((unsigned)(size_t)&As[0],
                             ctxb + (size_t)m0 * D_ + kk,
                             /*tile_w=*/32, /*tile_h=*/64, /*stride=*/D_);
        }
        const float* wp = w + (size_t)(n0 + brow) * D_ + kk + bcol;
        __builtin_prefetch(wp + 32, 0, 1);
#pragma unroll
        for (int j = 0; j < 16; ++j) Bs[brow * 32 + bcol + j] = cvt_bf16(wp[j]);
        if (wave == 0) __builtin_amdgcn_s_wait_tensorcnt(0);
        __syncthreads();
        v16bf b0 = load_b_frag_rowk(Bs + (nt + 0) * 16 * 32, 32);
        v16bf b1 = load_b_frag_rowk(Bs + (nt + 1) * 16 * 32, 32);
        v16bf a0 = load_a_frag(As + (mt + 0) * 16 * 32, 32);
        v16bf a1 = load_a_frag(As + (mt + 1) * 16 * 32, 32);
        acc00 = wmma_bf16(a0, b0, acc00);
        acc01 = wmma_bf16(a0, b1, acc01);
        acc10 = wmma_bf16(a1, b0, acc10);
        acc11 = wmma_bf16(a1, b1, acc11);
        __syncthreads();
    }
    auto store_tile = [&](const v8f& a, int tm, int tn) {
        const int n = n0 + tn * 16 + lr;
        const float bn = bias[n];
#pragma unroll
        for (int r = 0; r < 8; ++r) {
            int gm = m0 + tm * 16 + r + grp * 8;
            out[(size_t)gm * D_ + n] = mask[gm] ? (a[r] + bn) : 0.0f;
        }
    };
    store_tile(acc00, mt + 0, nt + 0);
    store_tile(acc01, mt + 0, nt + 1);
    store_tile(acc10, mt + 1, nt + 0);
    store_tile(acc11, mt + 1, nt + 1);
}

// ---------------------------------------------------------------------------
extern "C" void kernel_launch(void* const* d_in, const int* in_sizes, int n_in,
                              void* d_out, int out_size, void* d_ws, size_t ws_size,
                              hipStream_t stream) {
    const float*         q    = (const float*)d_in[0];
    const unsigned char* qmk  = (const unsigned char*)d_in[1];   // bool mask
    const unsigned char* kvmk = (const unsigned char*)d_in[2];   // bool mask
    const float*         ew   = (const float*)d_in[3];
    const float*         wq   = (const float*)d_in[4];
    const float*         wk   = (const float*)d_in[5];
    const float*         wv   = (const float*)d_in[6];
    const float*         wo   = (const float*)d_in[7];
    const float*         bo   = (const float*)d_in[8];
    float*               out  = (float*)d_out;

    const size_t NE = (size_t)B_ * H_ * S_ * HD_;   // 4,194,304 bf16 elems
    __bf16* Qh  = (__bf16*)d_ws;
    __bf16* Kh  = Qh + NE;
    __bf16* Vh  = Kh + NE;
    __bf16* ctx = Vh + NE;                          // [B*S, D] bf16

    dim3 gProj(64, 8, 1);       // M/64 x N/128
    dim3 blk(256, 1, 1);
    proj_bf16_kernel<<<gProj, blk, 0, stream>>>(q, wq, qmk, Qh);
    proj_bf16_kernel<<<gProj, blk, 0, stream>>>(q, wk, kvmk, Kh);
    proj_bf16_kernel<<<gProj, blk, 0, stream>>>(q, wv, kvmk, Vh);

    dim3 gAttn(S_ / 16, H_, B_);
    attn_kernel<<<gAttn, blk, 0, stream>>>(Qh, Kh, Vh, ew, qmk, kvmk, ctx);

    out_proj_kernel<<<gProj, blk, 0, stream>>>(ctx, wo, bo, qmk, out);
}
